// MultiHeadSelfAttention_56684978372644
// MI455X (gfx1250) — matrix operand order repaired and checked by reading, not compiled, not run on hardware
//
#include <hip/hip_runtime.h>

// ---------------------------------------------------------------------------
// MHA block for MI455X (gfx1250): bf16 WMMA GEMMs + async-to-LDS tile staging
// + fp32 flash softmax + fused RoPE.   B=2, T=2048, C=2048, H=16, hd=128
// ---------------------------------------------------------------------------

#define B_   2
#define T_   2048
#define C_   2048
#define H_   16
#define HD_  128
#define BT_  (B_ * T_)        // 4096
#define F_   (3 * C_)         // 6144

typedef __bf16 bf16_t;
typedef __attribute__((ext_vector_type(16))) __bf16          bf16x16;
typedef __attribute__((ext_vector_type(8)))  float           f32x8;
typedef __attribute__((ext_vector_type(4)))  float           f32x4;
typedef __attribute__((ext_vector_type(16))) unsigned short  u16x16;
typedef __attribute__((ext_vector_type(8)))  unsigned short  u16x8;
typedef __attribute__((ext_vector_type(4)))  unsigned short  u16x4;

union ABFrag {
    bf16x16 bf;
    u16x16  u;
    u16x8   h[2];
};

__device__ __forceinline__ unsigned short f32_to_bf16(float f) {
    unsigned int u = __float_as_uint(f);
    u += 0x7FFFu + ((u >> 16) & 1u);     // round-to-nearest-even
    return (unsigned short)(u >> 16);
}

__device__ __forceinline__ f32x8 wmma_bf16(bf16x16 a, bf16x16 b, f32x8 c) {
    return __builtin_amdgcn_wmma_f32_16x16x32_bf16(
        /*neg_a=*/false, a, /*neg_b=*/false, b,
        /*c_mod=*/(short)0, c, /*reuse_a=*/false, /*reuse_b=*/false);
}

__device__ __forceinline__ f32x8 zero8() {
    f32x8 z = {0.f, 0.f, 0.f, 0.f, 0.f, 0.f, 0.f, 0.f};
    return z;
}

// CDNA5 async memory->LDS copy (ASYNCcnt-tracked, no VGPR staging).
// lds_off = low 32 bits of generic pointer to __shared__ (LDS aperture keeps
// addr[31:0] as the LDS byte address).
__device__ __forceinline__ void async_ld_b128(unsigned lds_off,
                                              const unsigned short* g) {
    asm volatile("global_load_async_to_lds_b128 %0, %1, off"
                 :: "v"(lds_off), "v"(g) : "memory");
}
__device__ __forceinline__ void wait_async0() {
    asm volatile("s_wait_asynccnt 0x0" ::: "memory");
}
__device__ __forceinline__ unsigned lds_addr(const void* p) {
    return (unsigned)(unsigned long long)p;
}

// ---------------------------------------------------------------------------
// Kernel 1: fp32 -> bf16 conversion (vectorized, memory bound)
// ---------------------------------------------------------------------------
__global__ void cvt_bf16_4(const f32x4* __restrict__ src,
                           u16x4* __restrict__ dst, int n4) {
    int i = blockIdx.x * blockDim.x + threadIdx.x;
    if (i < n4) {
        f32x4 f = src[i];
        u16x4 o;
#pragma unroll
        for (int e = 0; e < 4; e++) o[e] = f32_to_bf16(f[e]);
        dst[i] = o;
    }
}

// ---------------------------------------------------------------------------
// Kernel 2: QKV GEMM (X[4096x2048] * Wqkv[6144x2048]^T), async double-buffered
// tile staging, fused RoPE epilogue. Writes Qh,Kh bf16 [b,h,t,d], V bf16
// transposed [b,h,d,t]. Tile 128x128 per 256-thread block (8 waves).
// ---------------------------------------------------------------------------
__global__ __launch_bounds__(256) void qkv_gemm_rope(
    const unsigned short* __restrict__ X,   // bf16 [BT_][C_]
    const unsigned short* __restrict__ W,   // bf16 [F_][C_]
    unsigned short* __restrict__ Qh,
    unsigned short* __restrict__ Kh,
    unsigned short* __restrict__ Vt)
{
    __shared__ alignas(32) unsigned short As[2][128][32];
    __shared__ alignas(32) unsigned short Bs[2][128][32];

    const int tid  = threadIdx.x;
    const int lane = tid & 31;
    const int w    = tid >> 5;     // 0..7
    const int wm   = w >> 2;       // 0..1
    const int wn   = w & 3;        // 0..3
    const int m0   = blockIdx.y * 128;
    const int n0   = blockIdx.x * 128;
    const int rlo  = lane & 15;
    const int hi   = lane >> 4;

    f32x8 acc[4][2];
#pragma unroll
    for (int i = 0; i < 4; i++)
#pragma unroll
        for (int j = 0; j < 2; j++) acc[i][j] = zero8();

    const int lr = tid >> 2;          // 0..63
    const int lk = (tid & 3) * 8;     // 0,8,16,24
    const unsigned short* gA0 = &X[(size_t)(m0 + lr) * C_ + lk];
    const unsigned short* gA1 = &X[(size_t)(m0 + 64 + lr) * C_ + lk];
    const unsigned short* gB0 = &W[(size_t)(n0 + lr) * C_ + lk];
    const unsigned short* gB1 = &W[(size_t)(n0 + 64 + lr) * C_ + lk];

    // stage 0 prefetch
    async_ld_b128(lds_addr(&As[0][lr][lk]),      gA0);
    async_ld_b128(lds_addr(&As[0][64 + lr][lk]), gA1);
    async_ld_b128(lds_addr(&Bs[0][lr][lk]),      gB0);
    async_ld_b128(lds_addr(&Bs[0][64 + lr][lk]), gB1);

    const int NK = C_ / 32;   // 64
    for (int it = 0; it < NK; ++it) {
        wait_async0();
        __syncthreads();
        if (it + 1 < NK) {
            const int k1 = (it + 1) * 32;
            const int nb = (it + 1) & 1;
            async_ld_b128(lds_addr(&As[nb][lr][lk]),      gA0 + k1);
            async_ld_b128(lds_addr(&As[nb][64 + lr][lk]), gA1 + k1);
            async_ld_b128(lds_addr(&Bs[nb][lr][lk]),      gB0 + k1);
            async_ld_b128(lds_addr(&Bs[nb][64 + lr][lk]), gB1 + k1);
        }
        const int buf = it & 1;

        ABFrag a[4], b[2];
#pragma unroll
        for (int mf = 0; mf < 4; mf++) {
            int row = wm * 64 + mf * 16 + rlo;
            int kb  = hi * 16;
            a[mf].h[0] = *(const u16x8*)&As[buf][row][kb];
            a[mf].h[1] = *(const u16x8*)&As[buf][row][kb + 8];
        }
#pragma unroll
        for (int nf = 0; nf < 2; nf++) {
            int col = wn * 32 + nf * 16 + rlo;
            b[nf].u = *(const u16x16*)&Bs[buf][col][hi * 16];
        }
#pragma unroll
        for (int mf = 0; mf < 4; mf++)
#pragma unroll
            for (int nf = 0; nf < 2; nf++)
                acc[mf][nf] = wmma_bf16(a[mf].bf, b[nf].bf, acc[mf][nf]);
    }

    // Epilogue: tile (128 cols, 128-aligned) lies in exactly one (part, head).
    const int part = n0 >> 11;            // 0=q 1=k 2=v
    const int head = (n0 & 2047) >> 7;

#pragma unroll
    for (int mf = 0; mf < 4; mf++) {
#pragma unroll
        for (int nf = 0; nf < 2; nf++) {
            const int d = wn * 32 + nf * 16 + rlo;   // 0..127
#pragma unroll
            for (int i = 0; i < 8; i++) {
                float v = acc[mf][nf][i];
                int r  = m0 + wm * 64 + mf * 16 + i + hi * 8;
                int bb = r >> 11;
                int t  = r & 2047;
                int bh = bb * H_ + head;
                if (part == 2) {
                    Vt[(bh * HD_ + d) * T_ + t] = f32_to_bf16(v);
                } else {
                    // RoPE: pair (even,odd) d lives in adjacent lanes
                    float partner = __shfl_xor(v, 1, 32);
                    int   ip   = d >> 1;
                    float freq = __expf(-((float)(2 * ip) * (1.0f / 128.0f)) *
                                        9.210340371976184f);   // ln(10000)
                    float ang = (float)t * freq;
                    float s, c;
                    __sincosf(ang, &s, &c);
                    float rv = (d & 1) ? (partner * s + v * c)
                                       : (v * c - partner * s);
                    unsigned short bv = f32_to_bf16(rv);
                    if (part == 0) Qh[(bh * T_ + t) * HD_ + d] = bv;
                    else           Kh[(bh * T_ + t) * HD_ + d] = bv;
                }
            }
        }
    }
}

// ---------------------------------------------------------------------------
// Kernel 3: causal flash attention. Block = (b,h, 64 q-rows), 4 waves.
// Each wave owns 16 q-rows; streams K/V (L2-resident) in 32-key steps.
// ---------------------------------------------------------------------------
__global__ __launch_bounds__(128) void flash_attn(
    const unsigned short* __restrict__ Qh,
    const unsigned short* __restrict__ Kh,
    const unsigned short* __restrict__ Vt,
    unsigned short* __restrict__ O)     // bf16 [BT_][C_]
{
    __shared__ alignas(32) unsigned short Ps[4][16][32];   // per-wave P tile

    const int lane = threadIdx.x & 31;
    const int w    = threadIdx.x >> 5;
    const int bh   = blockIdx.y;               // 0..31
    const int qw   = blockIdx.x * 64 + w * 16; // wave q base
    const int rlo  = lane & 15;
    const int hi   = lane >> 4;

    const unsigned short* Qb = Qh + (size_t)bh * T_ * HD_;
    const unsigned short* Kb = Kh + (size_t)bh * T_ * HD_;
    const unsigned short* Vb = Vt + (size_t)bh * HD_ * T_;

    // Q A-fragments (16 rows x 128 d = 4 frags of K=32), persistent
    ABFrag qf[4];
#pragma unroll
    for (int dc = 0; dc < 4; dc++) {
        int row = qw + rlo;
        int kb  = dc * 32 + hi * 16;
        qf[dc].h[0] = *(const u16x8*)&Qb[row * HD_ + kb];
        qf[dc].h[1] = *(const u16x8*)&Qb[row * HD_ + kb + 8];
    }

    float m[8], l[8];
    f32x8 o[8];
#pragma unroll
    for (int i = 0; i < 8; i++) { m[i] = -3.0e38f; l[i] = 0.f; o[i] = zero8(); }

    const float scale = 0.08838834764831845f;   // 1/sqrt(128)
    const int jmax = (qw + 15) >> 5;

    for (int j = 0; j <= jmax; j++) {
        const int kbase = j << 5;

        // S = Q K^T  (16 x 32 keys): two C frags, contract over d=128
        f32x8 s0 = zero8(), s1 = zero8();
#pragma unroll
        for (int dc = 0; dc < 4; dc++) {
            ABFrag b0, b1;
            int kb = dc * 32 + hi * 16;
            b0.u = *(const u16x16*)&Kb[(kbase + rlo) * HD_ + kb];
            b1.u = *(const u16x16*)&Kb[(kbase + 16 + rlo) * HD_ + kb];
            s0 = wmma_bf16(qf[dc].bf, b0.bf, s0);
            s1 = wmma_bf16(qf[dc].bf, b1.bf, s1);
        }

        const bool need_mask = (kbase + 31) > qw;   // wave-uniform
#pragma unroll
        for (int i = 0; i < 8; i++) { s0[i] *= scale; s1[i] *= scale; }
        if (need_mask) {
#pragma unroll
            for (int i = 0; i < 8; i++) {
                int q = qw + i + hi * 8;
                s0[i] = (kbase + rlo      > q) ? -3.0e38f : s0[i];
                s1[i] = (kbase + 16 + rlo > q) ? -3.0e38f : s1[i];
            }
        }

        // online softmax per row (rows map to 16-lane groups)
        float p0[8], p1[8], alpha[8];
#pragma unroll
        for (int i = 0; i < 8; i++) {
            float mx = fmaxf(s0[i], s1[i]);
#pragma unroll
            for (int off = 1; off < 16; off <<= 1)
                mx = fmaxf(mx, __shfl_xor(mx, off, 32));
            float mn = fmaxf(m[i], mx);
            alpha[i] = __expf(m[i] - mn);
            p0[i]    = __expf(s0[i] - mn);
            p1[i]    = __expf(s1[i] - mn);
            float rs = p0[i] + p1[i];
#pragma unroll
            for (int off = 1; off < 16; off <<= 1)
                rs += __shfl_xor(rs, off, 32);
            l[i] = l[i] * alpha[i] + rs;
            m[i] = mn;
        }
#pragma unroll
        for (int dc = 0; dc < 8; dc++)
#pragma unroll
            for (int i = 0; i < 8; i++) o[dc][i] *= alpha[i];

        // transpose P (C-layout) -> A-layout via per-wave LDS
#pragma unroll
        for (int i = 0; i < 8; i++) {
            int row = i + hi * 8;
            Ps[w][row][rlo]      = f32_to_bf16(p0[i]);
            Ps[w][row][16 + rlo] = f32_to_bf16(p1[i]);
        }
        ABFrag pa;
        {
            int kb = hi * 16;
            pa.h[0] = *(const u16x8*)&Ps[w][rlo][kb];
            pa.h[1] = *(const u16x8*)&Ps[w][rlo][kb + 8];
        }

        // O += P * V  (contract 32 keys; V transposed so loads are contiguous)
#pragma unroll
        for (int dc = 0; dc < 8; dc++) {
            ABFrag vb;
            vb.u = *(const u16x16*)&Vb[(dc * 16 + rlo) * T_ + kbase + hi * 16];
            o[dc] = wmma_bf16(pa.bf, vb.bf, o[dc]);
        }
    }

    // normalize + store bf16 [b,t, h*128+d]
    const int bb = bh >> 4, head = bh & 15;
#pragma unroll
    for (int i = 0; i < 8; i++) {
        float inv = 1.0f / l[i];
        int q = qw + i + hi * 8;
#pragma unroll
        for (int dc = 0; dc < 8; dc++) {
            int d = dc * 16 + rlo;
            O[((size_t)(bb * T_ + q)) * C_ + head * HD_ + d] =
                f32_to_bf16(o[dc][i] * inv);
        }
    }
}

// ---------------------------------------------------------------------------
// Kernel 4: output projection (attn[4096x2048] * Wproj[2048x2048]^T) -> fp32
// Same async double-buffered tile engine.
// ---------------------------------------------------------------------------
__global__ __launch_bounds__(256) void proj_gemm(
    const unsigned short* __restrict__ Xa,  // bf16 [BT_][C_]
    const unsigned short* __restrict__ W,   // bf16 [C_][C_]
    float* __restrict__ out)                // fp32 [BT_][C_]
{
    __shared__ alignas(32) unsigned short As[2][128][32];
    __shared__ alignas(32) unsigned short Bs[2][128][32];

    const int tid  = threadIdx.x;
    const int lane = tid & 31;
    const int w    = tid >> 5;
    const int wm   = w >> 2;
    const int wn   = w & 3;
    const int m0   = blockIdx.y * 128;
    const int n0   = blockIdx.x * 128;
    const int rlo  = lane & 15;
    const int hi   = lane >> 4;

    f32x8 acc[4][2];
#pragma unroll
    for (int i = 0; i < 4; i++)
#pragma unroll
        for (int j = 0; j < 2; j++) acc[i][j] = zero8();

    const int lr = tid >> 2;
    const int lk = (tid & 3) * 8;
    const unsigned short* gA0 = &Xa[(size_t)(m0 + lr) * C_ + lk];
    const unsigned short* gA1 = &Xa[(size_t)(m0 + 64 + lr) * C_ + lk];
    const unsigned short* gB0 = &W[(size_t)(n0 + lr) * C_ + lk];
    const unsigned short* gB1 = &W[(size_t)(n0 + 64 + lr) * C_ + lk];

    async_ld_b128(lds_addr(&As[0][lr][lk]),      gA0);
    async_ld_b128(lds_addr(&As[0][64 + lr][lk]), gA1);
    async_ld_b128(lds_addr(&Bs[0][lr][lk]),      gB0);
    async_ld_b128(lds_addr(&Bs[0][64 + lr][lk]), gB1);

    const int NK = C_ / 32;
    for (int it = 0; it < NK; ++it) {
        wait_async0();
        __syncthreads();
        if (it + 1 < NK) {
            const int k1 = (it + 1) * 32;
            const int nb = (it + 1) & 1;
            async_ld_b128(lds_addr(&As[nb][lr][lk]),      gA0 + k1);
            async_ld_b128(lds_addr(&As[nb][64 + lr][lk]), gA1 + k1);
            async_ld_b128(lds_addr(&Bs[nb][lr][lk]),      gB0 + k1);
            async_ld_b128(lds_addr(&Bs[nb][64 + lr][lk]), gB1 + k1);
        }
        const int buf = it & 1;

        ABFrag a[4], b[2];
#pragma unroll
        for (int mf = 0; mf < 4; mf++) {
            int row = wm * 64 + mf * 16 + rlo;
            int kb  = hi * 16;
            a[mf].h[0] = *(const u16x8*)&As[buf][row][kb];
            a[mf].h[1] = *(const u16x8*)&As[buf][row][kb + 8];
        }
#pragma unroll
        for (int nf = 0; nf < 2; nf++) {
            int col = wn * 32 + nf * 16 + rlo;
            b[nf].u = *(const u16x16*)&Bs[buf][col][hi * 16];
        }
#pragma unroll
        for (int mf = 0; mf < 4; mf++)
#pragma unroll
            for (int nf = 0; nf < 2; nf++)
                acc[mf][nf] = wmma_bf16(a[mf].bf, b[nf].bf, acc[mf][nf]);
    }

#pragma unroll
    for (int mf = 0; mf < 4; mf++)
#pragma unroll
        for (int nf = 0; nf < 2; nf++) {
            int col = n0 + wn * 32 + nf * 16 + rlo;
#pragma unroll
            for (int i = 0; i < 8; i++) {
                int r = m0 + wm * 64 + mf * 16 + i + hi * 8;
                out[(size_t)r * C_ + col] = acc[mf][nf][i];
            }
        }
}

// ---------------------------------------------------------------------------
// Launch
// ---------------------------------------------------------------------------
extern "C" void kernel_launch(void* const* d_in, const int* in_sizes, int n_in,
                              void* d_out, int out_size, void* d_ws, size_t ws_size,
                              hipStream_t stream) {
    const float* x      = (const float*)d_in[0];   // [B,T,C]
    const float* w_qkv  = (const float*)d_in[1];   // [3C,C]
    const float* w_proj = (const float*)d_in[2];   // [C,C]
    float*       out    = (float*)d_out;           // [B,T,C] fp32

    char* ws = (char*)d_ws;
    unsigned short* xh    = (unsigned short*)(ws);             // 16 MB
    unsigned short* wqkvh = (unsigned short*)(ws + 16777216);  // 24 MB
    unsigned short* wph   = (unsigned short*)(ws + 41943040);  //  8 MB
    unsigned short* Qh    = (unsigned short*)(ws + 50331648);  // 16 MB
    unsigned short* Kh    = (unsigned short*)(ws + 67108864);  // 16 MB
    unsigned short* Vt    = (unsigned short*)(ws + 83886080);  // 16 MB
    unsigned short* attn  = (unsigned short*)(ws + 100663296); // 16 MB

    // 1) fp32 -> bf16 conversions
    int n4;
    n4 = BT_ * C_ / 4;
    cvt_bf16_4<<<(n4 + 255) / 256, 256, 0, stream>>>((const f32x4*)x, (u16x4*)xh, n4);
    n4 = F_ * C_ / 4;
    cvt_bf16_4<<<(n4 + 255) / 256, 256, 0, stream>>>((const f32x4*)w_qkv, (u16x4*)wqkvh, n4);
    n4 = C_ * C_ / 4;
    cvt_bf16_4<<<(n4 + 255) / 256, 256, 0, stream>>>((const f32x4*)w_proj, (u16x4*)wph, n4);

    // 2) QKV projection + RoPE + head-layout scatter
    qkv_gemm_rope<<<dim3(F_ / 128, BT_ / 128), 256, 0, stream>>>(xh, wqkvh, Qh, Kh, Vt);

    // 3) causal flash attention
    flash_attn<<<dim3(T_ / 64, B_ * H_), 128, 0, stream>>>(Qh, Kh, Vt, attn);

    // 4) output projection (fp32 out)
    proj_gemm<<<dim3(C_ / 128, BT_ / 128), 256, 0, stream>>>(attn, wph, out);

    (void)in_sizes; (void)n_in; (void)out_size; (void)ws_size;
}
